// Graph_module_net_0_loss_2_18631568130082
// MI455X (gfx1250) — matrix operand order repaired and checked by reading, not compiled, not run on hardware
//
#include <hip/hip_runtime.h>
#include <hip/hip_bf16.h>

#define B_      4
#define N_      1024
#define CIN_    256
#define MID_    256
#define OUT_    256
#define H_      4
#define G_      4
#define CHILDS_ 16

typedef _Float16 half_t;
typedef __attribute__((ext_vector_type(16))) _Float16 v16h;
typedef __attribute__((ext_vector_type(8)))  float    v8f;

// ---------------------------------------------------------------------------
// WMMA fragment helpers (CDNA5 wave32 layouts, CDNA5 ISA §7.12.2)
// ---------------------------------------------------------------------------

// A fragment: 16x32 f16 tile at (row0,k0) of row-major [M x K] matrix.
// Lane L holds row M = L&15; K = 16*(L>>4) + i.
__device__ inline v16h load_a_frag(const half_t* __restrict__ A, int lda,
                                   int row0, int k0, int lane) {
    int r = lane & 15, hf = lane >> 4;
    const half_t* p = A + (size_t)(row0 + r) * lda + k0 + hf * 16;
    v16h a;
#pragma unroll
    for (int i = 0; i < 16; ++i) a[i] = p[i];
    return a;
}

// B fragment from N-major (N x K row-major, "Bt") storage: contiguous in K.
// Lane L holds column N = L&15; K = 16*(L>>4) + i.
__device__ inline v16h load_b_frag_nk(const half_t* __restrict__ Bt, int ldk,
                                      int n0, int k0, int lane) {
    int c = lane & 15, hg = lane >> 4;
    const half_t* p = Bt + (size_t)(n0 + c) * ldk + k0 + hg * 16;
    v16h b;
#pragma unroll
    for (int i = 0; i < 16; ++i) b[i] = p[i];
    return b;
}

// B fragment from K-major (K x N row-major) storage: strided loads.
__device__ inline v16h load_b_frag_kn(const half_t* __restrict__ Bm, int ldn,
                                      int k0, int n0, int lane) {
    int c = lane & 15, hg = lane >> 4;
    const half_t* p = Bm + (size_t)(k0 + hg * 16) * ldn + n0 + c;
    v16h b;
#pragma unroll
    for (int i = 0; i < 16; ++i) b[i] = p[(size_t)i * ldn];
    return b;
}

__device__ inline v8f wmma_f16(v16h a, v16h b, v8f c) {
    return __builtin_amdgcn_wmma_f32_16x16x32_f16(false, a, false, b,
                                                  (short)0, c, false, false);
}

// ---------------------------------------------------------------------------
// Elementwise utilities
// ---------------------------------------------------------------------------

__global__ void k_cvt_f16(const float* __restrict__ s, half_t* __restrict__ d, int n) {
    int i = blockIdx.x * blockDim.x + threadIdx.x;
    if (i < n) d[i] = (half_t)s[i];
}

__global__ void k_zero(float* __restrict__ p, int n) {
    int i = blockIdx.x * blockDim.x + threadIdx.x;
    if (i < n) p[i] = 0.0f;
}

// ---------------------------------------------------------------------------
// gts = relu(gt_feat @ gt_w.T + gt_b)   M=4096, N=256, K=256
// Each wave: 16x64 output strip; loads for all 4 B-fragments issued before
// the 4 WMMAs so the compiler can use partial s_wait_loadcnt overlap.
// 1024 wave-jobs -> <<<128, 256>>>
// ---------------------------------------------------------------------------
__global__ void k_gemm_gts(const half_t* __restrict__ A,   // (4096 x 256) f16
                           const half_t* __restrict__ Bt,  // gt_w (256 x 256) = N x K
                           const float*  __restrict__ bias,
                           float*        __restrict__ C) { // (4096 x 256)
    int lane = threadIdx.x & 31;
    int wave = threadIdx.x >> 5;
    int job  = blockIdx.x * 8 + wave;         // 256 mT x 4 nStrips
    int m0 = (job >> 2) * 16, n0 = (job & 3) * 64;
    v8f acc[4] = {};
    for (int k0 = 0; k0 < CIN_; k0 += 32) {
        v16h a = load_a_frag(A, CIN_, m0, k0, lane);
        v16h bf[4];
#pragma unroll
        for (int t = 0; t < 4; ++t)
            bf[t] = load_b_frag_nk(Bt, CIN_, n0 + 16 * t, k0, lane);
#pragma unroll
        for (int t = 0; t < 4; ++t)
            acc[t] = wmma_f16(a, bf[t], acc[t]);
    }
    int col = lane & 15, hf = lane >> 4;
#pragma unroll
    for (int t = 0; t < 4; ++t) {
        float bs = bias[n0 + 16 * t + col];
#pragma unroll
        for (int v = 0; v < 8; ++v) {
            float r = acc[t][v] + bs;
            C[(size_t)(m0 + v + 8 * hf) * OUT_ + n0 + 16 * t + col] = r > 0.f ? r : 0.f;
        }
    }
}

// ---------------------------------------------------------------------------
// Grouped 1x1 conv: block-diag GEMM per (b,g): M=64(o) N=1024(n) K=64(i)
// x_is_nk==1: X is (B,N,CIN) row-major (layer1);  ==0: X is (B,256,N) (layer2)
// Each wave: 16x64 strip. 16*(4x16)=1024 jobs -> <<<128, 256>>>
// ---------------------------------------------------------------------------
__global__ void k_gconv(const half_t* __restrict__ W,     // (G,64,64)
                        const half_t* __restrict__ X,
                        const float*  __restrict__ bias,  // (256)
                        float*        __restrict__ Y,     // (B,256,1024)
                        half_t*       __restrict__ Yh,
                        int x_is_nk) {
    int lane = threadIdx.x & 31;
    int wave = threadIdx.x >> 5;
    int job  = blockIdx.x * 8 + wave;
    int bg  = job >> 6;
    int rem = job & 63;
    int b = bg >> 2, g = bg & 3;
    int m0 = (rem >> 4) * 16, n0 = (rem & 15) * 64;
    const half_t* Ap = W + (size_t)g * 64 * 64;
    v8f acc[4] = {};
    for (int k0 = 0; k0 < 64; k0 += 32) {
        v16h a = load_a_frag(Ap, 64, m0, k0, lane);
        v16h bf[4];
#pragma unroll
        for (int t = 0; t < 4; ++t) {
            if (x_is_nk) {  // Bt[n][k] = X[b][n][g*64+k]
                bf[t] = load_b_frag_nk(X + (size_t)b * N_ * CIN_ + g * 64, CIN_,
                                       n0 + 16 * t, k0, lane);
            } else {        // B[k][n] = X[b][g*64+k][n]
                bf[t] = load_b_frag_kn(X + ((size_t)b * MID_ + g * 64) * N_, N_,
                                       k0, n0 + 16 * t, lane);
            }
        }
#pragma unroll
        for (int t = 0; t < 4; ++t)
            acc[t] = wmma_f16(a, bf[t], acc[t]);
    }
    int col = lane & 15, hf = lane >> 4;
#pragma unroll
    for (int t = 0; t < 4; ++t) {
#pragma unroll
        for (int v = 0; v < 8; ++v) {
            int ch = g * 64 + m0 + v + 8 * hf;
            float r = acc[t][v] + bias[ch];
            r = r > 0.f ? r : 0.f;
            size_t idx = ((size_t)b * MID_ + ch) * N_ + n0 + 16 * t + col;
            Y[idx]  = r;
            Yh[idx] = (half_t)r;
        }
    }
}

// ---------------------------------------------------------------------------
// Big attention GEMM per (b,h): (64 x 1024) @ (1024 x 1024) ; B in N x K form
// Each wave: 16x64 strip, K-loop of 32: 1 A-frag + 4 B-frags -> 4 wmma.
// 16*(4x16)=1024 jobs -> <<<128, 256>>>
// ---------------------------------------------------------------------------
__global__ void k_gemm_attn(const half_t* __restrict__ Ah,   // (B,256,1024) f16
                            const half_t* __restrict__ AttT, // (B,H,N_i,N_j) f16
                            float*        __restrict__ O) {  // (B,256,1024)
    int lane = threadIdx.x & 31;
    int wave = threadIdx.x >> 5;
    int job  = blockIdx.x * 8 + wave;
    int bh  = job >> 6;
    int rem = job & 63;
    int b = bh >> 2, h = bh & 3;
    int m0 = (rem >> 4) * 16, n0 = (rem & 15) * 64;
    const half_t* Ap = Ah + ((size_t)b * MID_ + h * 64) * N_;
    const half_t* Bp = AttT + ((size_t)(b * H_ + h)) * N_ * N_;
    v8f acc[4] = {};
    for (int k0 = 0; k0 < N_; k0 += 32) {
        v16h a = load_a_frag(Ap, N_, m0, k0, lane);
        v16h bf[4];
#pragma unroll
        for (int t = 0; t < 4; ++t)
            bf[t] = load_b_frag_nk(Bp, N_, n0 + 16 * t, k0, lane);
#pragma unroll
        for (int t = 0; t < 4; ++t)
            acc[t] = wmma_f16(a, bf[t], acc[t]);
    }
    int col = lane & 15, hf = lane >> 4;
#pragma unroll
    for (int t = 0; t < 4; ++t) {
#pragma unroll
        for (int v = 0; v < 8; ++v) {
            int ch = h * 64 + m0 + v + 8 * hf;
            O[((size_t)b * MID_ + ch) * N_ + n0 + 16 * t + col] = acc[t][v];
        }
    }
}

// ---------------------------------------------------------------------------
// aj/ai head projections: aj[b,n,h] = feat[b,n,:]·W[h,0:C], ai uses W[h,C:2C]
// ---------------------------------------------------------------------------
__global__ void k_ajai(const float* __restrict__ feat,
                       long long sB, long long sN, long long sC,
                       const float* __restrict__ W, int C,
                       float* __restrict__ aj, float* __restrict__ ai) {
    int t = blockIdx.x * blockDim.x + threadIdx.x;
    if (t >= B_ * N_ * H_) return;
    int h = t & 3, n = (t >> 2) & (N_ - 1), b = t >> 12;
    const float* f  = feat + (long long)b * sB + (long long)n * sN;
    const float* wj = W + (size_t)h * 2 * C;
    const float* wi = wj + C;
    float sj = 0.f, si = 0.f;
    for (int c = 0; c < C; ++c) {
        float v = f[(long long)c * sC];
        sj += v * wj[c];
        si += v * wi[c];
    }
    aj[t] = sj;
    ai[t] = si;
}

// ---------------------------------------------------------------------------
// Top-k: one wave per (b,i,h) row; marks global present[j] (batch-wide quirk).
// attn_topk_input[b,i,j,h] = sigmoid(aj[b,j,h]+ai[b,i,h]+bias[h]) * roi[b,i,j]
// ---------------------------------------------------------------------------
__global__ void k_topk(const float* __restrict__ aj, const float* __restrict__ ai,
                       const float* __restrict__ bvec,
                       const float* __restrict__ masks_roi,
                       const float* __restrict__ score_mask,
                       float* __restrict__ present) {
    __shared__ float rowv[8][N_];
    int lane = threadIdx.x & 31;
    int w    = threadIdx.x >> 5;
    int row  = blockIdx.x * 8 + w;            // over B*N*H
    int h = row & 3, i = (row >> 2) & (N_ - 1), b = row >> 12;
    float aiv = ai[row] + bvec[h];
    for (int j = lane; j < N_; j += 32) {
        float ajv = aj[((b << 10) + j) * 4 + h];
        float s   = 1.f / (1.f + __expf(-(ajv + aiv)));
        float roi = masks_roi[((size_t)b << 20) + ((size_t)i << 10) + j] *
                    score_mask[(b << 10) + j];
        rowv[w][j] = s * roi;
    }
    __syncthreads();
    for (int it = 0; it < CHILDS_; ++it) {
        float bv = -1e30f; int bi = -1;
        for (int j = lane; j < N_; j += 32) {
            float v = rowv[w][j];
            if (v > bv) { bv = v; bi = j; }
        }
#pragma unroll
        for (int off = 16; off > 0; off >>= 1) {
            float ov = __shfl_xor(bv, off, 32);
            int   oi = __shfl_xor(bi, off, 32);
            if (ov > bv || (ov == bv && oi >= 0 && (bi < 0 || oi < bi))) { bv = ov; bi = oi; }
        }
        if (lane == 0 && bi >= 0) {
            present[bi]  = 1.0f;      // race-benign: all writers store 1.0
            rowv[w][bi]  = -1e30f;
        }
        __syncthreads();
    }
}

// ---------------------------------------------------------------------------
// Build a1/a2 directly in f16, (b,h,i,j) layout (= WMMA B-operand N x K):
// val = (sig*roi*roi*present[j] + f_mask(i==j, score_mask[b,i]==0)) / CHILDS
// ---------------------------------------------------------------------------
__global__ void k_attn_build(const float* __restrict__ aj, const float* __restrict__ ai,
                             const float* __restrict__ bvec,
                             const float* __restrict__ masks_roi,
                             const float* __restrict__ score_mask,
                             const float* __restrict__ present,
                             half_t* __restrict__ AttT) {
    int blk = blockIdx.x;                     // ((b*H)+h)*N + i
    int i = blk & (N_ - 1), h = (blk >> 10) & 3, b = blk >> 12;
    float aiv = ai[((b << 10) + i) * 4 + h] + bvec[h];
    float fm  = (score_mask[(b << 10) + i] == 0.f) ? 1.f : 0.f;
    half_t* out = AttT + ((size_t)blk << 10);
    const float inv = 1.0f / (float)CHILDS_;
    for (int j = threadIdx.x; j < N_; j += blockDim.x) {
        float ajv = aj[((b << 10) + j) * 4 + h];
        float s   = 1.f / (1.f + __expf(-(ajv + aiv)));
        float roi = masks_roi[((size_t)b << 20) + ((size_t)i << 10) + j] *
                    score_mask[(b << 10) + j];
        float v = s * roi * roi * present[j];
        if (i == j) v += fm;
        out[j] = (half_t)(v * inv);
    }
}

// ---------------------------------------------------------------------------
// LN over channels of o1m (B,256,1024) + residual into out1; emit f16 copy.
// ---------------------------------------------------------------------------
__global__ void k_ln_add(const float* __restrict__ om,
                         const float* __restrict__ gam, const float* __restrict__ bet,
                         float* __restrict__ out1, half_t* __restrict__ out1h) {
    int t = blockIdx.x * blockDim.x + threadIdx.x;   // over B*N
    if (t >= B_ * N_) return;
    int n = t & (N_ - 1), b = t >> 10;
    const float* src = om + (size_t)b * MID_ * N_ + n;
    float s = 0.f, ss = 0.f;
    for (int c = 0; c < MID_; ++c) {
        float v = src[(size_t)c * N_];
        s += v; ss += v * v;
    }
    float m  = s * (1.f / MID_);
    float vr = ss * (1.f / MID_) - m * m;
    float rs = rsqrtf(vr + 1e-6f);
    float* dst = out1 + (size_t)b * MID_ * N_ + n;
    for (int c = 0; c < MID_; ++c) {
        float v  = src[(size_t)c * N_];
        float ln = (v - m) * rs * gam[c] + bet[c];
        float nv = dst[(size_t)c * N_] + ln;
        dst[(size_t)c * N_] = nv;
        out1h[(size_t)b * MID_ * N_ + (size_t)c * N_ + n] = (half_t)nv;
    }
}

// ---------------------------------------------------------------------------
// Final LN: node_feat = LN(o2m^T); out_final[b,n,c] = out2[b,c,n] + node_feat
// ---------------------------------------------------------------------------
__global__ void k_ln_final(const float* __restrict__ om,   // (B,256,1024)
                           const float* __restrict__ gam, const float* __restrict__ bet,
                           const float* __restrict__ out2, // (B,256,1024)
                           float* __restrict__ out_final,  // (B,N,256)
                           float* __restrict__ node_out) { // (B,N,256)
    int t = blockIdx.x * blockDim.x + threadIdx.x;   // over B*N
    if (t >= B_ * N_) return;
    int n = t & (N_ - 1), b = t >> 10;
    const float* src = om + (size_t)b * OUT_ * N_ + n;
    float s = 0.f, ss = 0.f;
    for (int c = 0; c < OUT_; ++c) {
        float v = src[(size_t)c * N_];
        s += v; ss += v * v;
    }
    float m  = s * (1.f / OUT_);
    float vr = ss * (1.f / OUT_) - m * m;
    float rs = rsqrtf(vr + 1e-6f);
    size_t rowo = (size_t)t * OUT_;
    for (int c = 0; c < OUT_; ++c) {
        float v  = src[(size_t)c * N_];
        float nf = (v - m) * rs * gam[c] + bet[c];
        node_out[rowo + c]  = nf;
        out_final[rowo + c] = out2[(size_t)b * OUT_ * N_ + (size_t)c * N_ + n] + nf;
    }
}

// ---------------------------------------------------------------------------
// Host launch
// ---------------------------------------------------------------------------
extern "C" void kernel_launch(void* const* d_in, const int* in_sizes, int n_in,
                              void* d_out, int out_size, void* d_ws, size_t ws_size,
                              hipStream_t stream) {
    const float* x          = (const float*)d_in[0];
    const float* masks_roi  = (const float*)d_in[1];
    const float* score_mask = (const float*)d_in[2];
    const float* gt_feat    = (const float*)d_in[3];
    const float* W_att1     = (const float*)d_in[4];
    const float* b_att1     = (const float*)d_in[5];
    const float* W_att2     = (const float*)d_in[6];
    const float* b_att2     = (const float*)d_in[7];
    const float* conv1_w    = (const float*)d_in[8];
    const float* conv1_b    = (const float*)d_in[9];
    const float* conv2_w    = (const float*)d_in[10];
    const float* conv2_b    = (const float*)d_in[11];
    const float* ln1_g      = (const float*)d_in[12];
    const float* ln1_b      = (const float*)d_in[13];
    const float* ln2_g      = (const float*)d_in[14];
    const float* ln2_b      = (const float*)d_in[15];
    const float* gt_w       = (const float*)d_in[16];
    const float* gt_b       = (const float*)d_in[17];

    float* out2_dst = (float*)d_out;                       // (B,N,OUT)
    float* gts_dst  = (float*)d_out + (size_t)B_ * N_ * OUT_;
    float* node_dst = (float*)d_out + (size_t)2 * B_ * N_ * OUT_;

    // workspace carve-up
    char* w = (char*)d_ws;
    auto alloc = [&](size_t bytes) -> char* {
        char* p = w;
        w += (bytes + 255) & ~(size_t)255;
        return p;
    };
    const size_t nBNC = (size_t)B_ * N_ * CIN_;            // 1,048,576
    const size_t nAtt = (size_t)B_ * H_ * N_ * N_;         // 16,777,216

    half_t* xh    = (half_t*)alloc(nBNC * 2);
    half_t* gtfh  = (half_t*)alloc(nBNC * 2);
    half_t* gtwh  = (half_t*)alloc((size_t)OUT_ * CIN_ * 2);
    half_t* w1h   = (half_t*)alloc((size_t)G_ * 64 * 64 * 2);
    half_t* w2h   = (half_t*)alloc((size_t)G_ * 64 * 64 * 2);
    float*  aj1   = (float*)alloc((size_t)B_ * N_ * H_ * 4);
    float*  ai1   = (float*)alloc((size_t)B_ * N_ * H_ * 4);
    float*  aj2   = (float*)alloc((size_t)B_ * N_ * H_ * 4);
    float*  ai2   = (float*)alloc((size_t)B_ * N_ * H_ * 4);
    float*  pres  = (float*)alloc((size_t)2 * N_ * 4);     // present1 | present2
    float*  pres1 = pres;
    float*  pres2 = pres + N_;
    half_t* att1h = (half_t*)alloc(nAtt * 2);
    half_t* att2h = (half_t*)alloc(nAtt * 2);
    float*  out1  = (float*)alloc(nBNC * 4);               // (B,256,1024)
    half_t* out1h = (half_t*)alloc(nBNC * 2);
    float*  o1m   = (float*)alloc(nBNC * 4);
    float*  out2b = (float*)alloc(nBNC * 4);               // (B,256,1024)
    half_t* out2h = (half_t*)alloc(nBNC * 2);
    float*  o2m   = (float*)alloc(nBNC * 4);
    (void)ws_size; (void)in_sizes; (void)n_in; (void)out_size;

    // 1) precision conversions
    k_cvt_f16<<<(int)((nBNC + 255) / 256), 256, 0, stream>>>(x, xh, (int)nBNC);
    k_cvt_f16<<<(int)((nBNC + 255) / 256), 256, 0, stream>>>(gt_feat, gtfh, (int)nBNC);
    k_cvt_f16<<<256, 256, 0, stream>>>(gt_w, gtwh, OUT_ * CIN_);
    k_cvt_f16<<<64, 256, 0, stream>>>(conv1_w, w1h, G_ * 64 * 64);
    k_cvt_f16<<<64, 256, 0, stream>>>(conv2_w, w2h, G_ * 64 * 64);
    k_zero<<<8, 256, 0, stream>>>(pres, 2 * N_);

    // 2) gts = relu(gt_feat @ gt_w.T + gt_b)
    k_gemm_gts<<<128, 256, 0, stream>>>(gtfh, gtwh, gt_b, gts_dst);

    // 3) layer 1 attention projections: x is (B,N,C)
    k_ajai<<<64, 256, 0, stream>>>(x, (long long)N_ * CIN_, (long long)CIN_, 1LL,
                                   W_att1, CIN_, aj1, ai1);

    // 4) out1 = relu(gconv1(x^T))
    k_gconv<<<128, 256, 0, stream>>>(w1h, xh, conv1_b, out1, out1h, 1);

    // 5) attention 1: top-k present mask, then build a1 in f16
    k_topk<<<2048, 256, 0, stream>>>(aj1, ai1, b_att1, masks_roi, score_mask, pres1);
    k_attn_build<<<16384, 256, 0, stream>>>(aj1, ai1, b_att1, masks_roi, score_mask,
                                            pres1, att1h);

    // 6) o1m = out1 @ a1 per (b,h)
    k_gemm_attn<<<128, 256, 0, stream>>>(out1h, att1h, o1m);

    // 7) out1 += LN(o1m^T)^T ; refresh f16 copy
    k_ln_add<<<16, 256, 0, stream>>>(o1m, ln1_g, ln1_b, out1, out1h);

    // 8) layer 2 attention projections: out1 is channel-major (B,C,N)
    k_ajai<<<64, 256, 0, stream>>>(out1, (long long)MID_ * N_, 1LL, (long long)N_,
                                   W_att2, MID_, aj2, ai2);

    // 9) out2 = relu(gconv2(out1))
    k_gconv<<<128, 256, 0, stream>>>(w2h, out1h, conv2_b, out2b, out2h, 0);

    // 10) attention 2
    k_topk<<<2048, 256, 0, stream>>>(aj2, ai2, b_att2, masks_roi, score_mask, pres2);
    k_attn_build<<<16384, 256, 0, stream>>>(aj2, ai2, b_att2, masks_roi, score_mask,
                                            pres2, att2h);

    // 11) o2m = out2 @ a2 per (b,h)
    k_gemm_attn<<<128, 256, 0, stream>>>(out2h, att2h, o2m);

    // 12) node_feat = LN(o2m^T); out2_final = out2 + node_feat^T (transposed out)
    k_ln_final<<<16, 256, 0, stream>>>(o2m, ln2_g, ln2_b, out2b, out2_dst, node_dst);
}
